// BoundaryRepresentation_38216619000306
// MI455X (gfx1250) — matrix operand order repaired and checked by reading, not compiled, not run on hardware
//
#include <hip/hip_runtime.h>
#include <hip/hip_bf16.h>

// ---------------------------------------------------------------------------
// CDNA5 (gfx1250) WMMA types & helpers
// ---------------------------------------------------------------------------
typedef __attribute__((ext_vector_type(16))) __bf16 v16bf;
typedef __attribute__((ext_vector_type(8)))  __bf16 v8bf;
typedef __attribute__((ext_vector_type(8)))  float  v8f;

__device__ __forceinline__ __bf16 f2bf(float f) { return (__bf16)f; }
__device__ __forceinline__ float  bf2f(__bf16 h) { return (float)h; }

__device__ __forceinline__ v16bf cat16(v8bf lo, v8bf hi)
{
    return __builtin_shufflevector(lo, hi, 0, 1, 2, 3, 4, 5, 6, 7,
                                           8, 9, 10, 11, 12, 13, 14, 15);
}
__device__ __forceinline__ v8bf ldg8(const __bf16* p) { return *(const v8bf*)p; }

__device__ __forceinline__ v8f wmma_bf16(v16bf a, v16bf b, v8f c)
{
    return __builtin_amdgcn_wmma_f32_16x16x32_bf16(
               false, a, false, b, (short)0, c, false, false);
}

#define LL   256   // sequence length
#define DD1  768
#define DD2  128
#define GAPC 30
#define CKK  3200  // conv implicit-GEMM K = 5*5*128

// ---------------------------------------------------------------------------
// Generic batched GEMM on bf16 operand streams with 2x2 register tiling:
//   C[b] = act(Abf[b] @ WT[b]^T + bias)
// Abf: M x K row-major bf16.  WT: N x K row-major bf16 (i.e. W transposed).
// Each wave computes a 32x32 output tile (4 WMMAs / k-step, 8 b128 loads ->
// 2 loads per WMMA).  blockDim = 128 (4 waves) -> block covers 32m x 128n.
// Requires M%32==0, N%32==0 (grid covers N in 128 chunks, partial waves
// guarded), K%32==0.
// ---------------------------------------------------------------------------
__global__ void gemm_wmma_kernel(const __bf16* __restrict__ Abf,
                                 const __bf16* __restrict__ WT,
                                 const float* __restrict__ bias,
                                 float* __restrict__ C,
                                 __bf16* __restrict__ Cbf,
                                 int M, int N, int K, int relu,
                                 long a_bs, long w_bs, long c_bs)
{
    const int wave = threadIdx.x >> 5;
    const int lane = threadIdx.x & 31;
    const int half = lane >> 4;
    const int lm   = lane & 15;

    const int n0 = (blockIdx.x * 4 + wave) * 32;
    const int m0 = blockIdx.y * 32;
    if (n0 >= N || m0 >= M) return;

    const __bf16* arow0 = Abf + (long)blockIdx.z * a_bs + (long)(m0 + lm) * K;
    const __bf16* arow1 = arow0 + (long)16 * K;
    const __bf16* wrow0 = WT  + (long)blockIdx.z * w_bs + (long)(n0 + lm) * K;
    const __bf16* wrow1 = wrow0 + (long)16 * K;

    v8f acc00 = {}, acc01 = {}, acc10 = {}, acc11 = {};
    for (int k0 = 0; k0 < K; k0 += 32) {
        // A tiles: elems i=0..15 -> k = k0 + half*16 + i (contiguous in A row)
        v16bf a0 = cat16(ldg8(arow0 + k0 + half * 16),
                         ldg8(arow0 + k0 + half * 16 + 8));
        v16bf a1 = cat16(ldg8(arow1 + k0 + half * 16),
                         ldg8(arow1 + k0 + half * 16 + 8));
        // B tiles: elems i=0..15 -> k = k0 + half*16 + i (contiguous in WT row)
        v16bf b0 = cat16(ldg8(wrow0 + k0 + half * 16),
                         ldg8(wrow0 + k0 + half * 16 + 8));
        v16bf b1 = cat16(ldg8(wrow1 + k0 + half * 16),
                         ldg8(wrow1 + k0 + half * 16 + 8));
        acc00 = wmma_bf16(a0, b0, acc00);
        acc01 = wmma_bf16(a0, b1, acc01);
        acc10 = wmma_bf16(a1, b0, acc10);
        acc11 = wmma_bf16(a1, b1, acc11);
    }

#pragma unroll
    for (int mt = 0; mt < 2; ++mt) {
#pragma unroll
        for (int nt = 0; nt < 2; ++nt) {
            const v8f acc = (mt == 0) ? (nt == 0 ? acc00 : acc01)
                                      : (nt == 0 ? acc10 : acc11);
            const int col = n0 + nt * 16 + lm;
            const float bv = bias ? bias[col] : 0.0f;
#pragma unroll
            for (int r = 0; r < 8; ++r) {
                const int m = m0 + mt * 16 + r + half * 8;
                float v = acc[r] + bv;
                if (relu) v = fmaxf(v, 0.0f);
                if (C)   C  [(long)blockIdx.z * c_bs + (long)m * N + col] = v;
                if (Cbf) Cbf[(long)blockIdx.z * c_bs + (long)m * N + col] = f2bf(v);
            }
        }
    }
}

// ---------------------------------------------------------------------------
// Transpose + fp32->bf16:  dst[n*K + k] = bf16(src[k*N + n]);  K,N % 32 == 0.
// ---------------------------------------------------------------------------
__global__ void transpose_cvt_kernel(const float* __restrict__ src,
                                     __bf16* __restrict__ dst, int K, int N)
{
    __shared__ float tile[32][33];
    const int k0 = blockIdx.x * 32;
    const int n0 = blockIdx.y * 32;
    const int tx = threadIdx.x & 31;
    const int ty = threadIdx.x >> 5;          // 0..7
#pragma unroll
    for (int r = ty; r < 32; r += 8)
        tile[r][tx] = src[(long)(k0 + r) * N + (n0 + tx)];
    __syncthreads();
#pragma unroll
    for (int r = ty; r < 32; r += 8)
        dst[(long)(n0 + r) * K + (k0 + tx)] = f2bf(tile[tx][r]);
}

// ---------------------------------------------------------------------------
// fp32 -> bf16 elementwise (for H_c).
// ---------------------------------------------------------------------------
__global__ void cvt_bf16_kernel(const float* __restrict__ src,
                                __bf16* __restrict__ dst, int n)
{
    const int i = blockIdx.x * blockDim.x + threadIdx.x;
    if (i < n) dst[i] = f2bf(src[i]);
}

// ---------------------------------------------------------------------------
// Pairwise build + LayerNorm:  Mout[b,i,j,:] = LN(Bs[b,i]+Be[b,j]+pos[|i-j|])
// One wave per (b,i,j) row of 128 channels; wave32 shuffle reductions.
// ---------------------------------------------------------------------------
__global__ void pair_ln_kernel(const float* __restrict__ Bs,
                               const float* __restrict__ Be,
                               const float* __restrict__ pos,
                               const float* __restrict__ g,
                               const float* __restrict__ bt,
                               __bf16* __restrict__ Mout)
{
    const int row  = blockIdx.x * (blockDim.x >> 5) + (threadIdx.x >> 5);
    const int lane = threadIdx.x & 31;
    const int j = row & (LL - 1);
    const int i = (row >> 8) & (LL - 1);
    const int b = row >> 16;

    const float* ps = Bs + (long)(b * LL + i) * DD2;
    const float* pe = Be + (long)(b * LL + j) * DD2;
    int d = i - j; if (d < 0) d = -d;
    const float* pp = pos + (long)d * DD2;

    float x[4], sum = 0.f, sq = 0.f;
#pragma unroll
    for (int t = 0; t < 4; ++t) {
        const int c = lane * 4 + t;
        const float v = ps[c] + pe[c] + pp[c];
        x[t] = v; sum += v; sq += v * v;
    }
#pragma unroll
    for (int o = 16; o > 0; o >>= 1) {
        sum += __shfl_xor(sum, o, 32);
        sq  += __shfl_xor(sq,  o, 32);
    }
    const float mu  = sum * (1.0f / DD2);
    const float var = sq * (1.0f / DD2) - mu * mu;
    const float rs  = rsqrtf(var + 1e-12f);

    __bf16* out = Mout + (long)row * DD2;
#pragma unroll
    for (int t = 0; t < 4; ++t) {
        const int c = lane * 4 + t;
        out[c] = f2bf((x[t] - mu) * rs * g[c] + bt[c]);
    }
}

// ---------------------------------------------------------------------------
// Implicit-GEMM 5x5 SAME conv (NHWC, 128->128) fused with scoring MLP
// (128 -> relu 128 -> 1) and mask/tri epilogue, writing G[b,i,j].
// Block = 128 threads = 4 waves: one 16-pixel strip (fixed h=i, j=w0..w0+15).
// Each wave owns a 32-wide output-channel chunk (2 accumulators): per k-chunk
// the A tile (2 loads) feeds 2 WMMAs -> 3 loads/WMMA.  K = 25*128 = 3200.
// WcT is the conv weight transposed to (co, k) bf16; W1T is (out, in) bf16.
// ---------------------------------------------------------------------------
__global__ void conv_score_kernel(const __bf16* __restrict__ Mb,   // (B,L,L,128) bf16
                                  const __bf16* __restrict__ WcT,  // (128, 3200) bf16
                                  const float*  __restrict__ cb,   // (128)
                                  const __bf16* __restrict__ W1T,  // (128, 128) bf16
                                  const float*  __restrict__ b1,   // (128)
                                  const float*  __restrict__ w2,   // (128)
                                  const float*  __restrict__ b2,   // (1)
                                  const float*  __restrict__ masks,// (B,L)
                                  float* __restrict__ G)           // (B,L,L)
{
    __shared__ __bf16 ldsA[16 * DD2];   // conv output tile (pixel-major)
    __shared__ __bf16 ldsH[16 * DD2];   // relu hidden tile

    const int wave = threadIdx.x >> 5;  // 0..3
    const int lane = threadIdx.x & 31;
    const int half = lane >> 4;
    const int lm   = lane & 15;

    const int bidx = blockIdx.x;            // b*(L*16) + h*16 + wblk
    const int wblk = bidx & 15;
    const int h    = (bidx >> 4) & (LL - 1);
    const int b    = bidx >> 12;
    const int w0   = wblk * 16;

    const int co0 = wave * 32;
    const __bf16* wrow0 = WcT + (long)(co0 + lm) * CKK;
    const __bf16* wrow1 = wrow0 + (long)16 * CKK;
    v8f acc0 = {}, acc1 = {};

    for (int tap = 0; tap < 25; ++tap) {
        const int kh = tap / 5, kw = tap % 5;
        const int hh = h + kh - 2;
        const int ww = w0 + lm + kw - 2;           // this lane's pixel column
        const bool inb = (hh >= 0) && (hh < LL) && (ww >= 0) && (ww < LL);
        const __bf16* src = Mb + (((long)(b * LL + hh) * LL + ww) * DD2);
        const __bf16* wk0 = wrow0 + tap * DD2;
        const __bf16* wk1 = wrow1 + tap * DD2;

        for (int c0 = 0; c0 < DD2; c0 += 32) {
            v16bf a;
            if (inb) {
                a = cat16(ldg8(src + c0 + half * 16),
                          ldg8(src + c0 + half * 16 + 8));
            } else {
                v16bf z = {};
                a = z;
            }
            v16bf bm0 = cat16(ldg8(wk0 + c0 + half * 16),
                              ldg8(wk0 + c0 + half * 16 + 8));
            v16bf bm1 = cat16(ldg8(wk1 + c0 + half * 16),
                              ldg8(wk1 + c0 + half * 16 + 8));
            acc0 = wmma_bf16(a, bm0, acc0);
            acc1 = wmma_bf16(a, bm1, acc1);
        }
    }

    // conv bias + stage to LDS (bf16, pixel-major)
#pragma unroll
    for (int nt = 0; nt < 2; ++nt) {
        const v8f acc = nt == 0 ? acc0 : acc1;
        const int co = co0 + nt * 16 + lm;
        const float bv = cb[co];
#pragma unroll
        for (int r = 0; r < 8; ++r) {
            const int pix = r + half * 8;
            ldsA[pix * DD2 + co] = f2bf(acc[r] + bv);
        }
    }
    __syncthreads();

    // scoring MLP layer 1: 16x128 @ 128x128 + b1, relu (wave owns 32 hidden)
    {
        const int h0 = wave * 32;
        const __bf16* w1row0 = W1T + (long)(h0 + lm) * DD2;
        const __bf16* w1row1 = w1row0 + (long)16 * DD2;
        v8f hacc0 = {}, hacc1 = {};
#pragma unroll
        for (int k0 = 0; k0 < DD2; k0 += 32) {
            v16bf a = cat16(*(const v8bf*)&ldsA[lm * DD2 + k0 + half * 16],
                            *(const v8bf*)&ldsA[lm * DD2 + k0 + half * 16 + 8]);
            v16bf bm0 = cat16(ldg8(w1row0 + k0 + half * 16),
                              ldg8(w1row0 + k0 + half * 16 + 8));
            v16bf bm1 = cat16(ldg8(w1row1 + k0 + half * 16),
                              ldg8(w1row1 + k0 + half * 16 + 8));
            hacc0 = wmma_bf16(a, bm0, hacc0);
            hacc1 = wmma_bf16(a, bm1, hacc1);
        }
#pragma unroll
        for (int nt = 0; nt < 2; ++nt) {
            const v8f hacc = nt == 0 ? hacc0 : hacc1;
            const int hc = h0 + nt * 16 + lm;
            const float bv = b1[hc];
#pragma unroll
            for (int r = 0; r < 8; ++r) {
                const int pix = r + half * 8;
                ldsH[pix * DD2 + hc] = f2bf(fmaxf(hacc[r] + bv, 0.0f));
            }
        }
    }
    __syncthreads();

    // layer 2 (128 -> 1) + mask epilogue
    if (threadIdx.x < 16) {
        const int pix = threadIdx.x;
        float s = b2[0];
        for (int c = 0; c < DD2; ++c)
            s += bf2f(ldsH[pix * DD2 + c]) * w2[c];
        const int j = w0 + pix;
        const float tri = (h <= j && (j - h) <= GAPC) ? 1.0f : 0.0f;
        const float mm  = masks[b * LL + h] * masks[b * LL + j] * tri;
        G[((long)b * LL + h) * LL + j] = s - 10000.0f * (1.0f - mm);
    }
}

// ---------------------------------------------------------------------------
// Wave-per-row softmax over 256 elements with stride (handles transposed G_e).
// Row r = b*256 + q; input element e at Gin[b*65536 + q*rstride + e*estride].
// Output Pbf[r*256 + e] in bf16 (feeds the P@V WMMA GEMM directly).
// ---------------------------------------------------------------------------
__global__ void softmax_kernel(const float* __restrict__ Gin,
                               __bf16* __restrict__ Pbf,
                               long rstride, long estride)
{
    const int row  = blockIdx.x * (blockDim.x >> 5) + (threadIdx.x >> 5);
    const int lane = threadIdx.x & 31;
    const int q = row & (LL - 1);
    const int b = row >> 8;
    const float* src = Gin + (long)b * (LL * LL) + (long)q * rstride;

    float v[8], mx = -3.0e38f;
#pragma unroll
    for (int t = 0; t < 8; ++t) {
        const int e = t * 32 + lane;
        v[t] = src[(long)e * estride];
        mx = fmaxf(mx, v[t]);
    }
#pragma unroll
    for (int o = 16; o > 0; o >>= 1) mx = fmaxf(mx, __shfl_xor(mx, o, 32));
    float sum = 0.f;
#pragma unroll
    for (int t = 0; t < 8; ++t) { v[t] = __expf(v[t] - mx); sum += v[t]; }
#pragma unroll
    for (int o = 16; o > 0; o >>= 1) sum += __shfl_xor(sum, o, 32);
    const float inv = 1.0f / sum;

    __bf16* dst = Pbf + (long)row * LL;
#pragma unroll
    for (int t = 0; t < 8; ++t) dst[t * 32 + lane] = f2bf(v[t] * inv);
}

// ---------------------------------------------------------------------------
// Tiny CLS-head MLP: out[b] = relu(H_cls[b] @ w1 + b1) @ w2 + b2  (768->768->1)
// One 256-thread block per batch element.
// ---------------------------------------------------------------------------
__global__ void cls_mlp_kernel(const float* __restrict__ Hc,
                               const float* __restrict__ w1,
                               const float* __restrict__ b1,
                               const float* __restrict__ w2,
                               const float* __restrict__ b2,
                               float* __restrict__ out)
{
    __shared__ float red[256];
    const int b = blockIdx.x;
    const float* x = Hc + (long)b * DD1;
    float partial = 0.f;
    for (int hd = threadIdx.x; hd < DD1; hd += 256) {
        float acc = b1[hd];
        for (int k = 0; k < DD1; ++k)
            acc += x[k] * w1[(long)k * DD1 + hd];
        partial += fmaxf(acc, 0.f) * w2[hd];
    }
    red[threadIdx.x] = partial;
    __syncthreads();
    for (int s = 128; s > 0; s >>= 1) {
        if (threadIdx.x < s) red[threadIdx.x] += red[threadIdx.x + s];
        __syncthreads();
    }
    if (threadIdx.x == 0) out[b] = red[0] + b2[0];
}

// ---------------------------------------------------------------------------
// Host-side orchestration
// ---------------------------------------------------------------------------
static inline void launch_gemm(const __bf16* Abf, const __bf16* WT,
                               const float* bias, float* C, __bf16* Cbf,
                               int M, int N, int K, int relu,
                               long a_bs, long w_bs, long c_bs, int batches,
                               hipStream_t s)
{
    dim3 block(128);
    dim3 grid((N + 127) / 128, M / 32, batches);
    gemm_wmma_kernel<<<grid, block, 0, s>>>(Abf, WT, bias, C, Cbf,
                                            M, N, K, relu, a_bs, w_bs, c_bs);
}

static inline void launch_transpose(const float* src, __bf16* dst,
                                    int K, int N, hipStream_t s)
{
    dim3 block(256);
    dim3 grid(K / 32, N / 32);
    transpose_cvt_kernel<<<grid, block, 0, s>>>(src, dst, K, N);
}

extern "C" void kernel_launch(void* const* d_in, const int* in_sizes, int n_in,
                              void* d_out, int out_size, void* d_ws, size_t ws_size,
                              hipStream_t stream)
{
    const float* H_c      = (const float*)d_in[0];
    const float* H_cls    = (const float*)d_in[1];
    const float* masks    = (const float*)d_in[2];
    const float* be_s_w1  = (const float*)d_in[3];
    const float* be_s_b1  = (const float*)d_in[4];
    const float* be_s_w2  = (const float*)d_in[5];
    const float* be_s_b2  = (const float*)d_in[6];
    const float* be_e_w1  = (const float*)d_in[7];
    const float* be_e_b1  = (const float*)d_in[8];
    const float* be_e_w2  = (const float*)d_in[9];
    const float* be_e_b2  = (const float*)d_in[10];
    const float* ses_map_w= (const float*)d_in[11];
    const float* ses_map_b= (const float*)d_in[12];
    const float* ses_pos  = (const float*)d_in[13];
    const float* see_map_w= (const float*)d_in[14];
    const float* see_map_b= (const float*)d_in[15];
    const float* see_pos  = (const float*)d_in[16];
    const float* conv_s_w = (const float*)d_in[17];
    const float* conv_s_b = (const float*)d_in[18];
    const float* conv_e_w = (const float*)d_in[19];
    const float* conv_e_b = (const float*)d_in[20];
    const float* scs_s_w1 = (const float*)d_in[21];
    const float* scs_s_b1 = (const float*)d_in[22];
    const float* scs_s_w2 = (const float*)d_in[23];
    const float* scs_s_b2 = (const float*)d_in[24];
    const float* scs_c_w1 = (const float*)d_in[25];
    const float* scs_c_b1 = (const float*)d_in[26];
    const float* scs_c_w2 = (const float*)d_in[27];
    const float* scs_c_b2 = (const float*)d_in[28];
    const float* sce_s_w1 = (const float*)d_in[29];
    const float* sce_s_b1 = (const float*)d_in[30];
    const float* sce_s_w2 = (const float*)d_in[31];
    const float* sce_s_b2 = (const float*)d_in[32];
    const float* sce_c_w1 = (const float*)d_in[33];
    const float* sce_c_b1 = (const float*)d_in[34];
    const float* sce_c_w2 = (const float*)d_in[35];
    const float* sce_c_b2 = (const float*)d_in[36];
    const float* w2s_w    = (const float*)d_in[37];
    const float* w2s_b    = (const float*)d_in[38];
    const float* w2e_w    = (const float*)d_in[39];
    const float* w2e_b    = (const float*)d_in[40];
    const float* ses_ln_g = (const float*)d_in[41];
    const float* see_ln_g = (const float*)d_in[42];
    const float* ses_ln_b = (const float*)d_in[43];
    const float* see_ln_b = (const float*)d_in[44];

    const int Bn = 2;
    const int ML = Bn * LL;          // 512 rows for the folded-batch GEMMs

    // ---- workspace layout (256B aligned) ----
    char* ws = (char*)d_ws;
    size_t off = 0;
    auto take = [&](size_t bytes) {
        char* p = ws + off;
        off = (off + bytes + 255) & ~(size_t)255;
        return p;
    };
    __bf16* Hbf   = (__bf16*)take((size_t)ML * DD1 * 2);
    __bf16* Tbf   = (__bf16*)take((size_t)ML * DD1 * 2);
    __bf16* hBsbf = (__bf16*)take((size_t)ML * DD1 * 2);
    __bf16* hBebf = (__bf16*)take((size_t)ML * DD1 * 2);
    __bf16* WTbig = (__bf16*)take((size_t)DD1 * DD1 * 2);  // reused per GEMM
    __bf16* WTmap = (__bf16*)take((size_t)DD2 * DD1 * 2);  // (128 x 768)
    __bf16* WcT   = (__bf16*)take((size_t)DD2 * CKK * 2);  // (128 x 3200)
    __bf16* W1T   = (__bf16*)take((size_t)DD2 * DD2 * 2);  // (128 x 128)
    float*  Bsb   = (float*) take((size_t)ML * DD2 * 4);
    float*  Beb   = (float*) take((size_t)ML * DD2 * 4);
    float*  V     = (float*) take((size_t)ML * DD1 * 4);
    __bf16* VT    = (__bf16*)take((size_t)Bn * DD1 * LL * 2); // per-batch (768x256)
    __bf16* Pbf   = (__bf16*)take((size_t)Bn * LL * LL * 2);
    __bf16* Mbf   = (__bf16*)take((size_t)Bn * LL * LL * DD2 * 2);

    // ---- output layout (fp32, concatenated in return order) ----
    float* out    = (float*)d_out;
    float* o_Bs   = out;                       // (2,256,768)
    float* o_Be   = out + 393216;              // (2,256,768)
    float* o_Gs   = out + 786432;              // (2,256,256)
    float* o_Ge   = out + 917504;              // (2,256,256)
    float* o_qs_s = out + 1048576;             // (2,1)
    float* o_qs_e = out + 1048578;             // (2,1)

    const dim3 pb(256);
    const dim3 cb128(128);
    const int nH = ML * DD1;

    // ===== shared front-end MLPs (activations chained in bf16) =====
    cvt_bf16_kernel<<<(nH + 255) / 256, pb, 0, stream>>>(H_c, Hbf, nH);

    launch_transpose(be_s_w1, WTbig, DD1, DD1, stream);
    launch_gemm(Hbf, WTbig, be_s_b1, nullptr, Tbf,   ML, DD1, DD1, 1, 0, 0, 0, 1, stream);
    launch_transpose(be_s_w2, WTbig, DD1, DD1, stream);
    launch_gemm(Tbf, WTbig, be_s_b2, nullptr, hBsbf, ML, DD1, DD1, 0, 0, 0, 0, 1, stream);

    launch_transpose(be_e_w1, WTbig, DD1, DD1, stream);
    launch_gemm(Hbf, WTbig, be_e_b1, nullptr, Tbf,   ML, DD1, DD1, 1, 0, 0, 0, 1, stream);
    launch_transpose(be_e_w2, WTbig, DD1, DD1, stream);
    launch_gemm(Tbf, WTbig, be_e_b2, nullptr, hBebf, ML, DD1, DD1, 0, 0, 0, 0, 1, stream);

    // ===== branch S =====
    launch_transpose(ses_map_w, WTmap, DD1, DD2, stream);
    launch_gemm(hBsbf, WTmap, ses_map_b, Bsb, nullptr, ML, DD2, DD1, 0, 0, 0, 0, 1, stream);
    launch_gemm(hBebf, WTmap, ses_map_b, Beb, nullptr, ML, DD2, DD1, 0, 0, 0, 0, 1, stream);

    pair_ln_kernel<<<Bn * LL * LL / 8, pb, 0, stream>>>(Bsb, Beb, ses_pos,
                                                        ses_ln_g, ses_ln_b, Mbf);

    launch_transpose(conv_s_w, WcT, CKK, DD2, stream);
    launch_transpose(scs_s_w1, W1T, DD2, DD2, stream);
    conv_score_kernel<<<Bn * LL * 16, cb128, 0, stream>>>(Mbf, WcT, conv_s_b,
                                                          W1T, scs_s_b1, scs_s_w2,
                                                          scs_s_b2, masks, o_Gs);

    launch_transpose(w2s_w, WTbig, DD1, DD1, stream);
    launch_gemm(hBsbf, WTbig, w2s_b, V, nullptr, ML, DD1, DD1, 0, 0, 0, 0, 1, stream);
    launch_transpose(V,              VT,              LL, DD1, stream);  // batch 0
    launch_transpose(V + LL * DD1,   VT + DD1 * LL,   LL, DD1, stream);  // batch 1

    softmax_kernel<<<Bn * LL / 8, pb, 0, stream>>>(o_Gs, Pbf, (long)LL, 1L);
    launch_gemm(Pbf, VT, nullptr, o_Bs, nullptr, LL, DD1, LL, 0,
                (long)LL * LL, (long)DD1 * LL, (long)LL * DD1, Bn, stream);

    // ===== branch E =====
    launch_transpose(see_map_w, WTmap, DD1, DD2, stream);
    launch_gemm(hBsbf, WTmap, see_map_b, Bsb, nullptr, ML, DD2, DD1, 0, 0, 0, 0, 1, stream);
    launch_gemm(hBebf, WTmap, see_map_b, Beb, nullptr, ML, DD2, DD1, 0, 0, 0, 0, 1, stream);

    pair_ln_kernel<<<Bn * LL * LL / 8, pb, 0, stream>>>(Bsb, Beb, see_pos,
                                                        see_ln_g, see_ln_b, Mbf);

    launch_transpose(conv_e_w, WcT, CKK, DD2, stream);
    launch_transpose(sce_s_w1, W1T, DD2, DD2, stream);
    conv_score_kernel<<<Bn * LL * 16, cb128, 0, stream>>>(Mbf, WcT, conv_e_b,
                                                          W1T, sce_s_b1, sce_s_w2,
                                                          sce_s_b2, masks, o_Ge);

    launch_transpose(w2e_w, WTbig, DD1, DD1, stream);
    launch_gemm(hBebf, WTbig, w2e_b, V, nullptr, ML, DD1, DD1, 0, 0, 0, 0, 1, stream);
    launch_transpose(V,              VT,              LL, DD1, stream);  // batch 0
    launch_transpose(V + LL * DD1,   VT + DD1 * LL,   LL, DD1, stream);  // batch 1

    // softmax over swapaxes(G_e): row = column j, elements stride 256
    softmax_kernel<<<Bn * LL / 8, pb, 0, stream>>>(o_Ge, Pbf, 1L, (long)LL);
    launch_gemm(Pbf, VT, nullptr, o_Be, nullptr, LL, DD1, LL, 0,
                (long)LL * LL, (long)DD1 * LL, (long)LL * DD1, Bn, stream);

    // ===== CLS heads =====
    cls_mlp_kernel<<<Bn, pb, 0, stream>>>(H_cls, scs_c_w1, scs_c_b1,
                                          scs_c_w2, scs_c_b2, o_qs_s);
    cls_mlp_kernel<<<Bn, pb, 0, stream>>>(H_cls, sce_c_w1, sce_c_b1,
                                          sce_c_w2, sce_c_b2, o_qs_e);

    (void)in_sizes; (void)n_in; (void)out_size; (void)ws_size;
}
